// LinearAttention_50225347559864
// MI455X (gfx1250) — matrix operand order repaired and checked by reading, not compiled, not run on hardware
//
#include <hip/hip_runtime.h>
#include <cstddef>

typedef __attribute__((ext_vector_type(16))) __bf16 v16bf;
typedef __attribute__((ext_vector_type(8)))  float  v8f;
typedef __attribute__((ext_vector_type(4)))  unsigned int u32x4;

#define DEVI static __device__ __forceinline__

// ---------------- problem constants ----------------
constexpr int kB = 8, kC = 256, kM = 16384, kHeads = 16, kCout = 256;
constexpr int kMT = 64;            // spatial columns per LDS tile
constexpr int kChunks = 64;        // partial chunks per batch (== blocks per batch)
constexpr int kTilesPerBlock = 4;  // 64 blocks * 4 tiles * 64 cols = 16384 = M

// ---------------- workspace layout (bytes) ----------------
constexpr size_t WS_WQB  = 0;                                        // bf16 [256][256]
constexpr size_t WS_WKB  = WS_WQB + (size_t)256 * 256 * 2;           // bf16 [256][256]
constexpr size_t WS_AP   = WS_WKB + (size_t)256 * 256 * 2;           // f32 [B][64][H][16][16]
constexpr size_t WS_QSP  = WS_AP  + (size_t)kB * kChunks * kHeads * 256 * 4;
constexpr size_t WS_KSP  = WS_QSP + (size_t)kB * kChunks * kHeads * 16 * 4;
constexpr size_t WS_WEFF = WS_KSP + (size_t)kB * kChunks * kHeads * 16 * 4; // bf16 [B][256][256]

// ---------------- helpers ----------------
DEVI unsigned short f2bf(float x) {           // RNE f32 -> bf16
  unsigned u = __builtin_bit_cast(unsigned, x);
  u += 0x7FFFu + ((u >> 16) & 1u);
  return (unsigned short)(u >> 16);
}
DEVI unsigned pk2(float a, float b) {
  return (unsigned)f2bf(a) | ((unsigned)f2bf(b) << 16);
}
DEVI float feat(float x) {                    // elu(x)+1
  return x > 0.0f ? x + 1.0f : __expf(x);
}

union BfPack {
  struct { u32x4 lo, hi; } u;
  v16bf v;
};

// A-operand (16 rows x 32 K, bf16) for one K-block kb from a row-major bf16
// matrix with row stride 256.  Lane L: row = L%16, half = L/16,
// elements 0-7 -> K = kb*32 + 8*half + (0..7); 8-15 -> K = kb*32+16+8*half+(0..7).
DEVI v16bf load_wA(const unsigned short* __restrict__ w, int rowbase, int kb, int lane) {
  const int r = lane & 15, hf = lane >> 4;
  const unsigned short* p = w + (size_t)(rowbase + r) * 256 + kb * 32 + hf * 8;
  BfPack t;
  t.u.lo = *(const u32x4*)p;
  t.u.hi = *(const u32x4*)(p + 16);
  return t.v;
}

// B-operand (32 K x 16 N) from LDS x tile [256][kMT] bf16.
// Lane L: column = mb + L%16, half = L/16; elements 0-7 -> K = kb*32+8*half+(0..7),
// 8-15 -> K = kb*32+16+8*half+(0..7).
DEVI v16bf lds_xop(const unsigned short* sx, int kb, int mb, int lane) {
  const int n = mb + (lane & 15), hf = lane >> 4;
  v16bf r;
#pragma unroll
  for (int s = 0; s < 16; ++s) {
    const int k = kb * 32 + (s < 8 ? 8 * hf + s : 16 + 8 * hf + (s - 8));
    r[s] = __builtin_bit_cast(__bf16, sx[(size_t)k * kMT + n]);
  }
  return r;
}

// Stage one x tile [256 x kMT] f32->bf16 into LDS (float4-coalesced).
DEVI void stage_x(const float* __restrict__ xb, int m0, unsigned short* sx,
                  int tid, int nthr) {
  for (int i = tid; i < (kC * kMT) / 4; i += nthr) {
    const int c  = i >> 4;           // 16 float4 quads per row
    const int mp = (i & 15) << 2;
    const float4 v = *(const float4*)(xb + (size_t)c * kM + m0 + mp);
    uint2 p;
    p.x = pk2(v.x, v.y);
    p.y = pk2(v.z, v.w);
    *(uint2*)&sx[c * kMT + mp] = p;
  }
}

// Prefetch next x tile (256 rows x 256 B = two 128 B lines per row) into
// WGP$/L2 while the current tile is being computed (global_prefetch_b8,
// no counters, no VGPR cost).
DEVI void prefetch_x(const float* __restrict__ xb, int m0, int tid, int nthr) {
  for (int i = tid; i < 512; i += nthr) {
    const int c  = i >> 1;
    const int mo = (i & 1) << 5;     // 0 or 32 floats (128 B)
    __builtin_prefetch(xb + (size_t)c * kM + m0 + mo, 0, 1);
  }
}

// ===================================================================
// Pass 1: Q/K projection (bf16 WMMA, f32 acc) + feature map, accumulate
// per-(b,h) A = Q K^T (WMMA over spatial), qsum/ksum partials.
// Grid: (64 chunks, 8 batches).  Block = 256 threads = 8 waves
// (2 waves/SIMD -> 512-VGPR budget per wave); each wave owns TWO heads
// so all four weight A-operand sets stay register-resident (no spills).
// ===================================================================
__global__ __launch_bounds__(256, 1)
void la_pass1(const float* __restrict__ x,
              const unsigned short* __restrict__ wqb,
              const unsigned short* __restrict__ wkb,
              float* __restrict__ Ap,     // [B][64][H][256]
              float* __restrict__ qsp,    // [B][64][H][16]
              float* __restrict__ ksp) {
  __shared__ __align__(32) unsigned short sx[kC * kMT];     // 32 KB
  __shared__ __align__(32) unsigned short stq[16][16 * 32]; // [head][d][m] 16 KB
  __shared__ __align__(32) unsigned short stk[16][32 * 16]; // [head][m][e] 16 KB

  const int b    = blockIdx.y;
  const int cx   = blockIdx.x;
  const int tid  = threadIdx.x;
  const int wv_  = tid >> 5;   // wave index 0..7
  const int lane = tid & 31;
  const int hf   = lane >> 4;
  const int lr   = lane & 15;
  const int h0   = wv_ * 2;    // this wave's two heads
  const int h1   = h0 + 1;

  const float* xb = x + (size_t)b * kC * kM;

  // weights for both heads resident as A-operands (8 K-blocks each)
  v16bf wqA0[8], wkA0[8], wqA1[8], wkA1[8];
#pragma unroll
  for (int kb = 0; kb < 8; ++kb) {
    wqA0[kb] = load_wA(wqb, h0 * 16, kb, lane);
    wkA0[kb] = load_wA(wkb, h0 * 16, kb, lane);
    wqA1[kb] = load_wA(wqb, h1 * 16, kb, lane);
    wkA1[kb] = load_wA(wkb, h1 * 16, kb, lane);
  }

  v8f Areg0, Areg1;
  float qp0[8], kp0[8], qp1[8], kp1[8];
#pragma unroll
  for (int g = 0; g < 8; ++g) {
    Areg0[g] = 0.0f; Areg1[g] = 0.0f;
    qp0[g] = 0.0f; kp0[g] = 0.0f; qp1[g] = 0.0f; kp1[g] = 0.0f;
  }

  for (int t = 0; t < kTilesPerBlock; ++t) {
    const int m0 = (cx + t * kChunks) * kMT;
    __syncthreads();                    // protect sx from previous iteration readers
    stage_x(xb, m0, sx, tid, 256);
    __syncthreads();
    if (t + 1 < kTilesPerBlock)         // hide HBM latency under the WMMAs below
      prefetch_x(xb, (cx + (t + 1) * kChunks) * kMT, tid, 256);

#pragma unroll
    for (int ch = 0; ch < 2; ++ch) {    // two 32-column chunks per tile
#pragma unroll
      for (int sub = 0; sub < 2; ++sub) {  // two 16-column WMMA N-tiles
        const int mb = ch * 32 + sub * 16;
        v8f qa0, ka0, qa1, ka1;
#pragma unroll
        for (int g = 0; g < 8; ++g) { qa0[g] = 0.0f; ka0[g] = 0.0f;
                                      qa1[g] = 0.0f; ka1[g] = 0.0f; }
#pragma unroll
        for (int kb = 0; kb < 8; ++kb) {
          const v16bf bx = lds_xop(sx, kb, mb, lane);
          qa0 = __builtin_amdgcn_wmma_f32_16x16x32_bf16(false, wqA0[kb], false, bx,
                                                        (short)0, qa0, false, false);
          ka0 = __builtin_amdgcn_wmma_f32_16x16x32_bf16(false, wkA0[kb], false, bx,
                                                        (short)0, ka0, false, false);
          qa1 = __builtin_amdgcn_wmma_f32_16x16x32_bf16(false, wqA1[kb], false, bx,
                                                        (short)0, qa1, false, false);
          ka1 = __builtin_amdgcn_wmma_f32_16x16x32_bf16(false, wkA1[kb], false, bx,
                                                        (short)0, ka1, false, false);
        }
        // feature map + running per-lane row sums
#pragma unroll
        for (int g = 0; g < 8; ++g) {
          qa0[g] = feat(qa0[g]);  qp0[g] += qa0[g];
          ka0[g] = feat(ka0[g]);  kp0[g] += ka0[g];
          qa1[g] = feat(qa1[g]);  qp1[g] += qa1[g];
          ka1[g] = feat(ka1[g]);  kp1[g] += ka1[g];
        }
        // stage K tiles as [m][e] (packed b128 store); Q tiles as [d][m]
        const int ml = sub * 16 + lr;   // m row within the 32-chunk
        u32x4 kv0, kv1;
#pragma unroll
        for (int g = 0; g < 4; ++g) {
          kv0[g] = pk2(ka0[2 * g], ka0[2 * g + 1]);
          kv1[g] = pk2(ka1[2 * g], ka1[2 * g + 1]);
        }
        *(u32x4*)&stk[h0][ml * 16 + hf * 8] = kv0;
        *(u32x4*)&stk[h1][ml * 16 + hf * 8] = kv1;
#pragma unroll
        for (int g = 0; g < 8; ++g) {
          stq[h0][(g + 8 * hf) * 32 + ml] = f2bf(qa0[g]);
          stq[h1][(g + 8 * hf) * 32 + ml] = f2bf(qa1[g]);
        }
      }
      // A += Q[16 x 32m] * K^T[32m x 16]  (per-wave LDS; DS in-order per wave)
      BfPack aq0, aq1;
      aq0.u.lo = *(const u32x4*)&stq[h0][lr * 32 + hf * 8];
      aq0.u.hi = *(const u32x4*)&stq[h0][lr * 32 + 16 + hf * 8];
      aq1.u.lo = *(const u32x4*)&stq[h1][lr * 32 + hf * 8];
      aq1.u.hi = *(const u32x4*)&stq[h1][lr * 32 + 16 + hf * 8];
      v16bf bk0, bk1;
#pragma unroll
      for (int s = 0; s < 16; ++s) {
        const int mk = s < 8 ? 8 * hf + s : 16 + 8 * hf + (s - 8);
        bk0[s] = __builtin_bit_cast(__bf16, stk[h0][mk * 16 + lr]);
        bk1[s] = __builtin_bit_cast(__bf16, stk[h1][mk * 16 + lr]);
      }
      Areg0 = __builtin_amdgcn_wmma_f32_16x16x32_bf16(false, aq0.v, false, bk0,
                                                      (short)0, Areg0, false, false);
      Areg1 = __builtin_amdgcn_wmma_f32_16x16x32_bf16(false, aq1.v, false, bk1,
                                                      (short)0, Areg1, false, false);
    }
  }

  // deterministic partial outputs (no atomics)
  float* ab0 = Ap + (size_t)((b * kChunks + cx) * kHeads + h0) * 256;
  float* ab1 = Ap + (size_t)((b * kChunks + cx) * kHeads + h1) * 256;
#pragma unroll
  for (int g = 0; g < 8; ++g) {
    ab0[(g + 8 * hf) * 16 + lr] = Areg0[g];
    ab1[(g + 8 * hf) * 16 + lr] = Areg1[g];
  }

#pragma unroll
  for (int g = 0; g < 8; ++g) {
#pragma unroll
    for (int off = 1; off < 16; off <<= 1) {
      qp0[g] += __shfl_xor(qp0[g], off, 32);
      kp0[g] += __shfl_xor(kp0[g], off, 32);
      qp1[g] += __shfl_xor(qp1[g], off, 32);
      kp1[g] += __shfl_xor(kp1[g], off, 32);
    }
  }
  if (lr == 0) {
    const size_t s0 = (size_t)((b * kChunks + cx) * kHeads + h0) * 16;
    const size_t s1 = (size_t)((b * kChunks + cx) * kHeads + h1) * 16;
#pragma unroll
    for (int g = 0; g < 8; ++g) {
      qsp[s0 + g + 8 * hf] = qp0[g];
      ksp[s0 + g + 8 * hf] = kp0[g];
      qsp[s1 + g + 8 * hf] = qp1[g];
      ksp[s1 + g + 8 * hf] = kp1[g];
    }
  }
}

// ===================================================================
// Pass 1.5: reduce partials, z[b,d] = 1/(qsum.ksum + 1e-6)  (reference
// broadcasting quirk: z indexes within-head channel d), fold into
// Weff[b][h*16+d][c] = z[b,d] * sum_e A[b,h,d,e] * Wv[h*16+e][c]  (bf16).
// Grid: (16 heads, 8 batches), 256 threads.
// ===================================================================
__global__ __launch_bounds__(256, 2)
void la_pass15(const float* __restrict__ wvw,
               const float* __restrict__ Ap,
               const float* __restrict__ qsp,
               const float* __restrict__ ksp,
               unsigned short* __restrict__ weff) {
  const int b = blockIdx.y, h = blockIdx.x, t = threadIdx.x;
  __shared__ float sA[256];
  __shared__ float sq[256], sk[256];   // [head][d] totals
  __shared__ float zs[16];

  float a = 0.0f;
  for (int j = 0; j < kChunks; ++j)
    a += Ap[(size_t)((b * kChunks + j) * kHeads + h) * 256 + t];
  sA[t] = a;

  const int th = t >> 4, td = t & 15;  // t = head*16 + d
  float q = 0.0f, k = 0.0f;
  for (int j = 0; j < kChunks; ++j) {
    const size_t sb = (size_t)((b * kChunks + j) * kHeads + th) * 16 + td;
    q += qsp[sb];
    k += ksp[sb];
  }
  sq[t] = q; sk[t] = k;
  __syncthreads();

  if (t < 16) {
    float s = 0.0f;
#pragma unroll
    for (int d = 0; d < 16; ++d) s += sq[t * 16 + d] * sk[t * 16 + d];
    zs[t] = 1.0f / (s + 1e-6f);
  }
  __syncthreads();

  const int c = t;
#pragma unroll
  for (int d = 0; d < 16; ++d) {
    float acc = 0.0f;
#pragma unroll
    for (int e = 0; e < 16; ++e)
      acc += sA[d * 16 + e] * wvw[(size_t)(h * 16 + e) * 256 + c];
    weff[((size_t)b * kCout + h * 16 + d) * 256 + c] = f2bf(acc * zs[d]);
  }
}

// ===================================================================
// Pass 2: out[b] = Weff[b] (256x256 bf16) * X[b] (256xM) -> f32.
// Block 512 (one weight set per wave, ~100 VGPRs < 256 cap).
// Output stores are non-temporal (written once, never re-read) so they
// do not evict x lines from L2.
// ===================================================================
__global__ __launch_bounds__(512, 1)
void la_pass2(const float* __restrict__ x,
              const unsigned short* __restrict__ weff,
              float* __restrict__ out) {
  __shared__ __align__(32) unsigned short sx[kC * kMT];     // 32 KB

  const int b    = blockIdx.y;
  const int cx   = blockIdx.x;
  const int tid  = threadIdx.x;
  const int wv_  = tid >> 5;
  const int lane = tid & 31;
  const int hf   = lane >> 4;
  const int lr   = lane & 15;

  const float* xb = x + (size_t)b * kC * kM;
  const unsigned short* wb = weff + (size_t)b * kCout * 256;

  v16bf wA[8];
#pragma unroll
  for (int kb = 0; kb < 8; ++kb)
    wA[kb] = load_wA(wb, wv_ * 16, kb, lane);

  for (int t = 0; t < kTilesPerBlock; ++t) {
    const int m0 = (cx + t * kChunks) * kMT;
    __syncthreads();
    stage_x(xb, m0, sx, tid, 512);
    __syncthreads();
    if (t + 1 < kTilesPerBlock)
      prefetch_x(xb, (cx + (t + 1) * kChunks) * kMT, tid, 512);

#pragma unroll
    for (int mc = 0; mc < 4; ++mc) {   // four 16-column N-tiles
      v8f acc;
#pragma unroll
      for (int g = 0; g < 8; ++g) acc[g] = 0.0f;
#pragma unroll
      for (int kb = 0; kb < 8; ++kb) {
        const v16bf bx = lds_xop(sx, kb, mc * 16, lane);
        acc = __builtin_amdgcn_wmma_f32_16x16x32_bf16(false, wA[kb], false, bx,
                                                      (short)0, acc, false, false);
      }
      const int m = m0 + mc * 16 + lr;
#pragma unroll
      for (int g = 0; g < 8; ++g) {
        const int o = wv_ * 16 + g + 8 * hf;
        __builtin_nontemporal_store(acc[g], &out[((size_t)b * kCout + o) * kM + m]);
      }
    }
  }
}

// ---------------- prep: weights f32 -> bf16 ----------------
__global__ __launch_bounds__(256)
void la_prep(const float* __restrict__ wq, const float* __restrict__ wk,
             unsigned short* __restrict__ wqb, unsigned short* __restrict__ wkb) {
  const int i = blockIdx.x * 256 + threadIdx.x;
  if (i < 256 * 256) {
    wqb[i] = f2bf(wq[i]);
    wkb[i] = f2bf(wk[i]);
  }
}

// ===================================================================
extern "C" void kernel_launch(void* const* d_in, const int* in_sizes, int n_in,
                              void* d_out, int out_size, void* d_ws, size_t ws_size,
                              hipStream_t stream) {
  (void)in_sizes; (void)n_in; (void)out_size; (void)ws_size;
  const float* x  = (const float*)d_in[0];
  const float* wq = (const float*)d_in[1];
  const float* wk = (const float*)d_in[2];
  const float* wv = (const float*)d_in[3];
  float* out = (float*)d_out;

  char* ws = (char*)d_ws;
  unsigned short* wqb  = (unsigned short*)(ws + WS_WQB);
  unsigned short* wkb  = (unsigned short*)(ws + WS_WKB);
  float*          Ap   = (float*)(ws + WS_AP);
  float*          qsp  = (float*)(ws + WS_QSP);
  float*          ksp  = (float*)(ws + WS_KSP);
  unsigned short* weff = (unsigned short*)(ws + WS_WEFF);

  la_prep<<<256, 256, 0, stream>>>(wq, wk, wqb, wkb);
  la_pass1<<<dim3(kChunks, kB), 256, 0, stream>>>(x, wqb, wkb, Ap, qsp, ksp);
  la_pass15<<<dim3(kHeads, kB), 256, 0, stream>>>(wv, Ap, qsp, ksp, weff);
  la_pass2<<<dim3(kChunks, kB), 512, 0, stream>>>(x, weff, out);
}
